// SimpleDeepseekV3MoE_11802570130393
// MI455X (gfx1250) — matrix operand order repaired and checked by reading, not compiled, not run on hardware
//
#include <hip/hip_runtime.h>
#include <hip/hip_bf16.h>

#ifndef __has_builtin
#define __has_builtin(x) 0
#endif

// ---------------------------------------------------------------------------
// MoE (DeepSeek-style top-2) for MI455X / gfx1250, wave32 + WMMA bf16.
//   B=2, S=2048 -> T=4096 tokens, D=1024, E=8, F=2048, TOP_K=2
// Routed compute only (mathematically identical to the dense-masked ref).
// ---------------------------------------------------------------------------

#define T_TOK 4096
#define DIM_D 1024
#define N_EXP 8
#define DIM_F 2048
#define CAP_E 4096          // max tokens per expert (top-2 => <= T)
#define TILE_M 32           // routed pairs per block tile
#define TILE_N 128          // output columns per block tile
#define MAX_TILES 128       // CAP_E / TILE_M
#define LDSTRIDE 40         // row stride in bf16 elems: 80 bytes, 16B aligned

typedef __bf16 bf16_t;
typedef __attribute__((ext_vector_type(16))) __bf16 v16bf;
typedef __attribute__((ext_vector_type(8)))  __bf16 v8bf;
typedef __attribute__((ext_vector_type(8)))  float  v8f;

static __device__ __forceinline__ v16bf cat8(v8bf lo, v8bf hi) {
    return __builtin_shufflevector(lo, hi,
        0, 1, 2, 3, 4, 5, 6, 7, 8, 9, 10, 11, 12, 13, 14, 15);
}

static __device__ __forceinline__ float fast_silu_mul(float g, float u) {
    // silu(g) * u with fast v_rcp_f32 instead of the IEEE divide expansion.
    return g * __builtin_amdgcn_rcpf(1.f + __expf(-g)) * u;
}

// Async DMA: 16 bytes global -> LDS, tracked by ASYNCcnt (CDNA5 ISA).
// VDST operand carries the workgroup-relative LDS byte address, which is the
// low 32 bits of a generic pointer into __shared__ (LDS aperture truncation).
static __device__ __forceinline__ void async_copy_b128(const void* gsrc,
                                                       void* ldst) {
    const unsigned long long ga = (unsigned long long)(uintptr_t)gsrc;
    const unsigned la = (unsigned)(uintptr_t)ldst;
    asm volatile("global_load_async_to_lds_b128 %0, %1, off"
                 :: "v"(la), "v"(ga) : "memory");
}

static __device__ __forceinline__ void wait_async_lds() {
#if __has_builtin(__builtin_amdgcn_s_wait_asynccnt)
    __builtin_amdgcn_s_wait_asynccnt(0);
#else
    asm volatile("s_wait_asynccnt 0x0" ::: "memory");
#endif
}

// ---------------------------------------------------------------------------
// Kernel 0: zero the output accumulator + expert counters.
// ---------------------------------------------------------------------------
__global__ void moe_zero_kernel(float4* __restrict__ out4, int n4,
                                int* __restrict__ counts) {
    int i = blockIdx.x * blockDim.x + threadIdx.x;
    if (i < n4) {
        float4 z; z.x = 0.f; z.y = 0.f; z.z = 0.f; z.w = 0.f;
        out4[i] = z;
    }
    if (blockIdx.x == 0 && threadIdx.x < N_EXP) counts[threadIdx.x] = 0;
}

// ---------------------------------------------------------------------------
// Kernel 1: routing. One wave32 per token. logits = x . gate_w[e], top-2,
// softmax over the two scores, scatter pair ids (2*t+k) into expert buckets.
// ---------------------------------------------------------------------------
__global__ __launch_bounds__(128)
void moe_route_kernel(const float* __restrict__ x,
                      const float* __restrict__ gw,
                      int* __restrict__ counts,
                      int* __restrict__ bucket,
                      float* __restrict__ pairw) {
    const int wave = threadIdx.x >> 5;
    const int lane = threadIdx.x & 31;
    const int t = blockIdx.x * 4 + wave;

    float acc[N_EXP];
#pragma unroll
    for (int e = 0; e < N_EXP; ++e) acc[e] = 0.f;

    for (int d = lane; d < DIM_D; d += 32) {
        const float xv = x[(size_t)t * DIM_D + d];
#pragma unroll
        for (int e = 0; e < N_EXP; ++e)
            acc[e] += xv * gw[(size_t)e * DIM_D + d];
    }
#pragma unroll
    for (int e = 0; e < N_EXP; ++e) {
#pragma unroll
        for (int off = 16; off > 0; off >>= 1)
            acc[e] += __shfl_xor(acc[e], off, 32);
    }

    if (lane == 0) {
        int e0 = 0; float v0 = acc[0];
#pragma unroll
        for (int e = 1; e < N_EXP; ++e)
            if (acc[e] > v0) { v0 = acc[e]; e0 = e; }
        int e1 = -1; float v1 = -3.4e38f;
#pragma unroll
        for (int e = 0; e < N_EXP; ++e)
            if (e != e0 && acc[e] > v1) { v1 = acc[e]; e1 = e; }
        // softmax over {v0, v1}, v0 >= v1
        const float z  = __expf(v1 - v0);
        const float w0 = 1.f / (1.f + z);
        const float w1 = z  / (1.f + z);
        const int p0 = atomicAdd(&counts[e0], 1);
        bucket[e0 * CAP_E + p0] = 2 * t;
        const int p1 = atomicAdd(&counts[e1], 1);
        bucket[e1 * CAP_E + p1] = 2 * t + 1;
        pairw[2 * t]     = w0;
        pairw[2 * t + 1] = w1;
    }
}

// ---------------------------------------------------------------------------
// Kernel 2: gate/up GEMMs + SwiGLU.  Block = 256 threads (8 waves).
// Tile: 32 routed pairs x 128 F-columns; K loop over D in steps of 32.
// Wave w owns M-half (w>>2) and a 32-col N slab (w&3): two 16x16 WMMA tiles
// per matrix product => 4 v_wmma per K-step per wave.
// ---------------------------------------------------------------------------
__global__ __launch_bounds__(256)
void moe_gateup_kernel(const float* __restrict__ x,
                       const float* __restrict__ wg,
                       const float* __restrict__ wu,
                       const int* __restrict__ counts,
                       const int* __restrict__ bucket,
                       bf16_t* __restrict__ hbuf) {
    __shared__ __attribute__((aligned(16))) bf16_t lA [TILE_M * LDSTRIDE];
    __shared__ __attribute__((aligned(16))) bf16_t lBg[TILE_N * LDSTRIDE];
    __shared__ __attribute__((aligned(16))) bf16_t lBu[TILE_N * LDSTRIDE];
    __shared__ int lPid[TILE_M];

    const int e    = blockIdx.x >> 7;          // / MAX_TILES
    const int tile = blockIdx.x & (MAX_TILES - 1);
    const int cnt  = counts[e];
    const int row0 = tile * TILE_M;
    if (row0 >= cnt) return;                   // block-uniform: EXEC stays full
    const int f0  = blockIdx.y * TILE_N;
    const int tid = threadIdx.x;

    if (tid < TILE_M) {
        int r = row0 + tid;
        if (r >= cnt) r = cnt - 1;             // clamp; stores guarded later
        lPid[tid] = bucket[e * CAP_E + r];
    }
    __syncthreads();

    const int lane = tid & 31;
    const int half = lane >> 4, ln = lane & 15;
    const int wv = tid >> 5;
    const int mh = wv >> 2;                    // M half: rows [mh*16, mh*16+16)
    const int ns = wv & 3;                     // N slab:  cols [ns*32, ns*32+32)

    v8f accg0 = {}, accg1 = {}, accu0 = {}, accu1 = {};

    for (int kb = 0; kb < DIM_D; kb += 32) {
        __syncthreads();
        // A tile: 32 pairs x 32 D, fp32 -> bf16
        for (int i = tid; i < TILE_M * 32; i += 256) {
            const int r = i >> 5, c = i & 31;
            const int tok = lPid[r] >> 1;
            lA[r * LDSTRIDE + c] = (bf16_t)x[(size_t)tok * DIM_D + kb + c];
        }
        // B tiles: 32 D x 128 F, stored transposed [f][d] for contiguous frags
        const float* wgp = wg + ((size_t)e * DIM_D + kb) * DIM_F + f0;
        const float* wup = wu + ((size_t)e * DIM_D + kb) * DIM_F + f0;
        for (int i = tid; i < 32 * TILE_N; i += 256) {
            const int d = i >> 7, f = i & 127;
            lBg[f * LDSTRIDE + d] = (bf16_t)wgp[(size_t)d * DIM_F + f];
            lBu[f * LDSTRIDE + d] = (bf16_t)wup[(size_t)d * DIM_F + f];
        }
        __syncthreads();

        // A fragment (16x32): lane<16 -> K 0..7 & 16..23; lane>=16 -> K 8..15 & 24..31
        const int ar = (mh * 16 + ln) * LDSTRIDE;
        v16bf a = cat8(*(const v8bf*)&lA[ar + half * 8],
                       *(const v8bf*)&lA[ar + 16 + half * 8]);
        // B fragments (32x16): lane<16 -> K 0..7 & 16..23; lane>=16 -> K 8..15 & 24..31
        const int n0 = (ns * 32 + ln) * LDSTRIDE + half * 8;
        const int n1 = n0 + 16 * LDSTRIDE;
        v16bf bg0 = cat8(*(const v8bf*)&lBg[n0], *(const v8bf*)&lBg[n0 + 16]);
        v16bf bg1 = cat8(*(const v8bf*)&lBg[n1], *(const v8bf*)&lBg[n1 + 16]);
        v16bf bu0 = cat8(*(const v8bf*)&lBu[n0], *(const v8bf*)&lBu[n0 + 16]);
        v16bf bu1 = cat8(*(const v8bf*)&lBu[n1], *(const v8bf*)&lBu[n1 + 16]);

        accg0 = __builtin_amdgcn_wmma_f32_16x16x32_bf16(
                    false, a, false, bg0, (short)0, accg0, false, false);
        accg1 = __builtin_amdgcn_wmma_f32_16x16x32_bf16(
                    false, a, false, bg1, (short)0, accg1, false, false);
        accu0 = __builtin_amdgcn_wmma_f32_16x16x32_bf16(
                    false, a, false, bu0, (short)0, accu0, false, false);
        accu1 = __builtin_amdgcn_wmma_f32_16x16x32_bf16(
                    false, a, false, bu1, (short)0, accu1, false, false);
    }

    // Epilogue: SwiGLU, store bf16 h. C layout: VGPR i -> M = i + 8*half, N = ln.
#pragma unroll
    for (int i = 0; i < 8; ++i) {
        const int R = mh * 16 + i + half * 8;
        if (row0 + R < cnt) {
            const int p = lPid[R];
            const size_t base = (size_t)p * DIM_F + f0 + ns * 32 + ln;
            hbuf[base]      = (bf16_t)fast_silu_mul(accg0[i], accu0[i]);
            hbuf[base + 16] = (bf16_t)fast_silu_mul(accg1[i], accu1[i]);
        }
    }
}

// ---------------------------------------------------------------------------
// Kernel 3: down GEMM + weighted scatter-add into the output.
// Tile: 32 pairs x 128 D-columns; K loop over F in steps of 32.
// A tile (bf16 h rows) is staged into LDS with GLOBAL_LOAD_ASYNC_TO_LDS_B128
// (ASYNCcnt-tracked DMA), overlapping the fp32->bf16 B-tile conversion.
// ---------------------------------------------------------------------------
__global__ __launch_bounds__(256)
void moe_down_kernel(const bf16_t* __restrict__ hbuf,
                     const float* __restrict__ wd,
                     const int* __restrict__ counts,
                     const int* __restrict__ bucket,
                     const float* __restrict__ pairw,
                     float* __restrict__ out) {
    __shared__ __attribute__((aligned(16))) bf16_t lA[TILE_M * LDSTRIDE];
    __shared__ __attribute__((aligned(16))) bf16_t lB[TILE_N * LDSTRIDE];
    __shared__ int   lPid[TILE_M];
    __shared__ float lW[TILE_M];

    const int e    = blockIdx.x >> 7;
    const int tile = blockIdx.x & (MAX_TILES - 1);
    const int cnt  = counts[e];
    const int row0 = tile * TILE_M;
    if (row0 >= cnt) return;
    const int d0  = blockIdx.y * TILE_N;
    const int tid = threadIdx.x;

    if (tid < TILE_M) {
        int r = row0 + tid;
        if (r >= cnt) r = cnt - 1;
        const int p = bucket[e * CAP_E + r];
        lPid[tid] = p;
        lW[tid]   = pairw[p];
    }
    __syncthreads();

    const int lane = tid & 31;
    const int half = lane >> 4, ln = lane & 15;
    const int wv = tid >> 5;
    const int mh = wv >> 2;
    const int ns = wv & 3;

    v8f acc0 = {}, acc1 = {};

    for (int kb = 0; kb < DIM_F; kb += 32) {
        __syncthreads();
        // A tile: 32 pairs x 32 F, already bf16 -> async DMA to LDS.
        // 32 rows x 64 B = 128 x 16 B chunks; lanes 0..127 move one b128 each
        // (waves 0..3 fully active -> EXEC uniform per wave).
        if (tid < 128) {
            const int r   = tid >> 2;          // pair row
            const int seg = tid & 3;           // 16-byte segment within row
            const bf16_t* gsrc =
                hbuf + (size_t)lPid[r] * DIM_F + kb + seg * 8;
            async_copy_b128(gsrc, &lA[r * LDSTRIDE + seg * 8]);
        }
        // B tile: 32 F x 128 D, fp32 -> bf16, transposed [d][f]
        const float* wdp = wd + ((size_t)e * DIM_F + kb) * DIM_D + d0;
        for (int i = tid; i < 32 * TILE_N; i += 256) {
            const int f = i >> 7, d = i & 127;
            lB[d * LDSTRIDE + f] = (bf16_t)wdp[(size_t)f * DIM_D + d];
        }
        wait_async_lds();
        __syncthreads();

        const int ar = (mh * 16 + ln) * LDSTRIDE;
        v16bf a = cat8(*(const v8bf*)&lA[ar + half * 8],
                       *(const v8bf*)&lA[ar + 16 + half * 8]);
        const int n0 = (ns * 32 + ln) * LDSTRIDE + half * 8;
        const int n1 = n0 + 16 * LDSTRIDE;
        v16bf b0 = cat8(*(const v8bf*)&lB[n0], *(const v8bf*)&lB[n0 + 16]);
        v16bf b1 = cat8(*(const v8bf*)&lB[n1], *(const v8bf*)&lB[n1 + 16]);

        acc0 = __builtin_amdgcn_wmma_f32_16x16x32_bf16(
                   false, a, false, b0, (short)0, acc0, false, false);
        acc1 = __builtin_amdgcn_wmma_f32_16x16x32_bf16(
                   false, a, false, b1, (short)0, acc1, false, false);
    }

#pragma unroll
    for (int i = 0; i < 8; ++i) {
        const int R = mh * 16 + i + half * 8;
        if (row0 + R < cnt) {
            const int p   = lPid[R];
            const int tok = p >> 1;
            const float w = lW[R];
            float* dst = &out[(size_t)tok * DIM_D + d0 + ns * 32 + ln];
            __hip_atomic_fetch_add(dst, acc0[i] * w, __ATOMIC_RELAXED,
                                   __HIP_MEMORY_SCOPE_AGENT);
            __hip_atomic_fetch_add(dst + 16, acc1[i] * w, __ATOMIC_RELAXED,
                                   __HIP_MEMORY_SCOPE_AGENT);
        }
    }
}

// ---------------------------------------------------------------------------
// Host-side launcher (graph-capture safe: no malloc/sync, all on `stream`).
// ---------------------------------------------------------------------------
extern "C" void kernel_launch(void* const* d_in, const int* in_sizes, int n_in,
                              void* d_out, int out_size, void* d_ws, size_t ws_size,
                              hipStream_t stream) {
    const float* x      = (const float*)d_in[0];  // [T, D]
    const float* gate_w = (const float*)d_in[1];  // [E, D]
    const float* w_gate = (const float*)d_in[2];  // [E, D, F]
    const float* w_up   = (const float*)d_in[3];  // [E, D, F]
    const float* w_down = (const float*)d_in[4];  // [E, F, D]
    float* out = (float*)d_out;                   // [T, D]

    // Workspace carve-up (all 256B aligned):
    //   counts : E ints
    //   bucket : E * CAP_E ints        (pair ids, pair = 2*t + k)
    //   pairw  : 2*T floats            (softmax routing weight per pair)
    //   hbuf   : 2*T * F bf16          (SwiGLU activations, ~33.6 MB)
    char* ws = (char*)d_ws;
    int*    counts = (int*)ws;
    int*    bucket = (int*)(ws + 256);
    float*  pairw  = (float*)(ws + 256 + (size_t)N_EXP * CAP_E * 4);
    bf16_t* hbuf   = (bf16_t*)(ws + 256 + (size_t)N_EXP * CAP_E * 4
                                   + (size_t)2 * T_TOK * 4);

    const int n4 = (T_TOK * DIM_D) / 4;
    moe_zero_kernel<<<(n4 + 255) / 256, 256, 0, stream>>>(
        (float4*)out, n4, counts);

    moe_route_kernel<<<T_TOK / 4, 128, 0, stream>>>(
        x, gate_w, counts, bucket, pairw);

    moe_gateup_kernel<<<dim3(N_EXP * MAX_TILES, DIM_F / TILE_N), 256, 0, stream>>>(
        x, w_gate, w_up, counts, bucket, hbuf);

    moe_down_kernel<<<dim3(N_EXP * MAX_TILES, DIM_D / TILE_N), 256, 0, stream>>>(
        hbuf, w_down, counts, bucket, pairw, out);
}
